// WorldEngine_28286654611820
// MI455X (gfx1250) — compile-verified, matrix operand order checked by reading, not run on hardware
//
#include <hip/hip_runtime.h>

// ---------------------------------------------------------------------------
// Fused GNN layer for MI455X (gfx1250, wave32, WMMA).
// Pipeline (all on `stream`, ordered by stream semantics):
//   1. init      : last=-1, attn_full=-inf, agg=0
//   2. xbf       : node_features -> bf16 (pair-packed dwords)
//   3. pack x4   : weights -> WMMA B-fragment-ready bf16 layout
//   4. last      : atomicMax per dst (torch "last write wins" edge)
//   5. q         : Q = X @ q_w + q_b  (per-node, scalar)
//   6. attn_sel  : kv + q.kv dot ONLY for selected edges (<=10K of 320K)
//   7. stats     : per-head max/sum over node axis
//   8. norm      : attn_norm = softmax over nodes
//   9. edge_msg  : persistent WMMA message MLP + attn scale + atomic agg
//  10. node_out  : persistent WMMA output MLP + residual + LayerNorm
// ---------------------------------------------------------------------------

#define NN 10000
#define NE 320000

typedef __attribute__((ext_vector_type(16))) __bf16 v16bf;
typedef __attribute__((ext_vector_type(8)))  float  v8f;
typedef __attribute__((ext_vector_type(4)))  int    v4i;

union Frag {
    v16bf v;
    unsigned int u[8];
    uint4 q[2];
};

__device__ __forceinline__ unsigned f2bf1(float f) {
    unsigned u = __float_as_uint(f);
    return (u + 0x7FFFu + ((u >> 16) & 1u)) >> 16;   // RNE fp32 -> bf16
}
__device__ __forceinline__ unsigned pack2bf(float a, float b) {
    return f2bf1(a) | (f2bf1(b) << 16);
}
__device__ __forceinline__ float gelu_exact(float x) {
    return 0.5f * x * (1.0f + erff(x * 0.70710678118654752f));
}
__device__ __forceinline__ void lds_fence() {
    // same-wave LDS is processed in order; this pins compiler ordering and
    // guarantees stores landed before dependent ds reads.
    asm volatile("s_wait_dscnt 0" ::: "memory");
}
__device__ __forceinline__ void async_wait0() {
#if __has_builtin(__builtin_amdgcn_s_wait_asynccnt)
    __builtin_amdgcn_s_wait_asynccnt(0);
#else
    asm volatile("s_wait_asynccnt 0" ::: "memory");
#endif
}
__device__ __forceinline__ v8f wmma_bf16(v16bf a, v16bf b, v8f c) {
    return __builtin_amdgcn_wmma_f32_16x16x32_bf16(
        false, a, false, b, (short)0, c, false, false);
}
__device__ __forceinline__ Frag load_frag_lds(const unsigned int* p) {
    Frag f;
    f.q[0] = *(const uint4*)(p);
    f.q[1] = *(const uint4*)(p + 4);
    return f;
}

// Stage ndw dwords global -> LDS, whole 256-thread block cooperating.
// Prefers gfx1250 GLOBAL_LOAD_ASYNC_TO_LDS_B128 (ASYNCcnt, no VGPR round trip).
__device__ __forceinline__ void stage_lds(unsigned int* dst, const unsigned int* src,
                                          int ndw, int tid) {
#if __has_builtin(__builtin_amdgcn_global_load_async_to_lds_b128)
    for (int i = tid * 4; i < ndw; i += 256 * 4)
        __builtin_amdgcn_global_load_async_to_lds_b128(
            (__attribute__((address_space(1))) v4i*)(src + i),
            (__attribute__((address_space(3))) v4i*)(dst + i),
            0, 0);
#else
    uint4* d4 = (uint4*)dst;
    const uint4* s4 = (const uint4*)src;
    for (int i = tid; i < (ndw >> 2); i += 256) d4[i] = s4[i];
#endif
}

// ---------------------------------------------------------------------------
// small prep kernels
// ---------------------------------------------------------------------------
__global__ void init_kernel(int* last, float* attn_full, float* agg) {
    int i = blockIdx.x * blockDim.x + threadIdx.x;
    int stride = gridDim.x * blockDim.x;
    for (int j = i; j < NN; j += stride) last[j] = -1;
    for (int j = i; j < NN * 8; j += stride) attn_full[j] = -__builtin_inff();
    for (int j = i; j < NN * 128; j += stride) agg[j] = 0.0f;
}

__global__ void xbf_kernel(const float* __restrict__ nf, unsigned int* __restrict__ Xbfu) {
    int i = blockIdx.x * blockDim.x + threadIdx.x;
    if (i < NN * 64) Xbfu[i] = pack2bf(nf[2 * i], nf[2 * i + 1]);
}

// Pack W[K x 128] (row-major, fan_in x fan_out) into B-fragment dwords:
// index = ((tk*8+tn)*32 + lane)*8 + r ; lane -> N, reg -> K (mirrors A layout).
__global__ void pack_weight_kernel(const float* __restrict__ W, unsigned int* __restrict__ out, int ntk) {
    int total = ntk * 8 * 256;
    int i = blockIdx.x * blockDim.x + threadIdx.x;
    if (i >= total) return;
    int r    = i & 7;
    int lane = (i >> 3) & 31;
    int tile = i >> 8;
    int tn = tile & 7, tk = tile >> 3;
    int Nc = tn * 16 + (lane & 15);
    int Kp = tk * 32 + ((r >> 2) << 4) + ((lane >> 4) << 3) + ((r & 3) << 1);
    out[i] = pack2bf(W[Kp * 128 + Nc], W[(Kp + 1) * 128 + Nc]);
}

__global__ void last_kernel(const int* __restrict__ dst, int* last) {
    int i = blockIdx.x * blockDim.x + threadIdx.x;
    if (i < NE) atomicMax(&last[dst[i]], i);
}

__global__ void q_kernel(const float* __restrict__ nf, const float* __restrict__ qw,
                         const float* __restrict__ qb, float* __restrict__ Q) {
    __shared__ float xs[128];
    int n = blockIdx.x, j = threadIdx.x;
    xs[j] = nf[n * 128 + j];
    __syncthreads();
    float a = qb[j];
    for (int k = 0; k < 128; ++k) a += xs[k] * qw[k * 128 + j];
    Q[n * 128 + j] = a;
}

// attention logits only for the single surviving edge per node
__global__ void attn_sel_kernel(const float* __restrict__ nf, const float* __restrict__ ea,
                                const int* __restrict__ srcI, const float* __restrict__ Q,
                                const float* __restrict__ kvw, const float* __restrict__ kvb,
                                const int* __restrict__ last, float* __restrict__ attn_full) {
    int n = blockIdx.x;
    int e = last[n];
    if (e < 0) return;
    __shared__ float sin_[192];
    __shared__ float sred[128];
    int j = threadIdx.x;
    int s = srcI[e];
    sin_[j] = nf[s * 128 + j];
    if (j < 64) sin_[128 + j] = ea[e * 64 + j];
    __syncthreads();
    float a = kvb[j];
    for (int k = 0; k < 192; ++k) a += sin_[k] * kvw[k * 128 + j];
    sred[j] = Q[n * 128 + j] * a;
    __syncthreads();
    if (j < 8) {
        float sum = 0.0f;
        for (int k = 0; k < 16; ++k) sum += sred[j * 16 + k];
        attn_full[n * 8 + j] = sum * 0.25f;   // HEAD_DIM^-0.5
    }
}

__global__ void softmax_stats_kernel(const float* __restrict__ attn_full, float* __restrict__ stats) {
    __shared__ float red[256];
    int h = blockIdx.x, tid = threadIdx.x;
    float m = -__builtin_inff();
    for (int n = tid; n < NN; n += 256) m = fmaxf(m, attn_full[n * 8 + h]);
    red[tid] = m; __syncthreads();
    for (int s = 128; s > 0; s >>= 1) { if (tid < s) red[tid] = fmaxf(red[tid], red[tid + s]); __syncthreads(); }
    float mx = red[0]; __syncthreads();
    float sum = 0.0f;
    if (mx > -1e30f)
        for (int n = tid; n < NN; n += 256) {
            float a = attn_full[n * 8 + h];
            if (a > -1e30f) sum += expf(a - mx);
        }
    red[tid] = sum; __syncthreads();
    for (int s = 128; s > 0; s >>= 1) { if (tid < s) red[tid] += red[tid + s]; __syncthreads(); }
    if (tid == 0) { stats[h * 2] = mx; stats[h * 2 + 1] = red[0]; }
}

__global__ void attn_norm_kernel(const float* __restrict__ attn_full, const float* __restrict__ stats,
                                 float* __restrict__ attn_norm) {
    int i = blockIdx.x * blockDim.x + threadIdx.x;
    if (i >= NN * 8) return;
    int h = i & 7;
    float a = attn_full[i];
    float mx = stats[h * 2], s = stats[h * 2 + 1];
    attn_norm[i] = (a > -1e30f && s > 0.0f) ? expf(a - mx) / s : 0.0f;
}

// ---------------------------------------------------------------------------
// Edge message MLP: persistent blocks, 16 edges per wave-iteration.
// LDS: msg_w1 frags (20480 dw) | msg_w2 frags (8192 dw) | 8 x 1184 dw scratch
// scratch: [0..15]=src [16..31]=dst [32..159]=attn_norm [160..1183]=m1 bf16
// ---------------------------------------------------------------------------
__global__ void __launch_bounds__(256)
edge_msg_kernel(const float* __restrict__ ea, const int* __restrict__ srcI, const int* __restrict__ dstI,
                const unsigned int* __restrict__ Xbfu, const float* __restrict__ attn_norm,
                const float* __restrict__ b1, const float* __restrict__ b2,
                const unsigned int* __restrict__ w1p, const unsigned int* __restrict__ w2p,
                float* __restrict__ agg) {
    extern __shared__ unsigned int smem[];
    unsigned int* ldsW1 = smem;
    unsigned int* ldsW2 = smem + 20480;
    const int tid = threadIdx.x;
    stage_lds(ldsW1, w1p, 20480, tid);   // staged ONCE per persistent block
    stage_lds(ldsW2, w2p, 8192, tid);
    async_wait0();
    __syncthreads();

    const int wave = tid >> 5;
    const int lane = tid & 31;
    const int M = lane & 15;
    const int half = lane >> 4;
    unsigned int* scr = smem + 28672 + wave * 1184;
    const int nt = NE / 16;              // 20000 tiles
    const int tstride = gridDim.x * 8;

    for (int t = blockIdx.x * 8 + wave; t < nt; t += tstride) {
        const int ebase = t * 16;

        // prefetch next tile's edge_attr rows while this tile computes
        {
            int tnext = t + tstride;
            if (tnext < nt && lane < 16)
                __builtin_prefetch(ea + (size_t)(tnext * 16 + lane) * 64, 0, 3);
        }

        if (lane < 16) {
            scr[lane]      = (unsigned)srcI[ebase + lane];
            scr[16 + lane] = (unsigned)dstI[ebase + lane];
        }
        lds_fence();
        for (int i = lane; i < 128; i += 32) {   // cache attn_norm[dst] per tile
            int row = i >> 3, h = i & 7;
            int dn = (int)scr[16 + row];
            scr[32 + i] = __float_as_uint(attn_norm[dn * 8 + h]);
        }
        const int srow = (int)scr[M];
        const int drow = (int)scr[16 + M];
        const int e = ebase + M;

        // GEMM1: m1 = [s|d|ea] (16x320) @ msg_w1 (320x128), bias-init accs
        v8f acc[8];
#pragma unroll
        for (int tn = 0; tn < 8; ++tn) {
            float bb = b1[tn * 16 + M];
#pragma unroll
            for (int i = 0; i < 8; ++i) acc[tn][i] = bb;
        }
#pragma unroll
        for (int kt = 0; kt < 10; ++kt) {
            Frag a;
#pragma unroll
            for (int r = 0; r < 8; ++r) {
                const int col = kt * 32 + ((r >> 2) << 4) + (half << 3) + ((r & 3) << 1);
                unsigned v;
                if (col < 128)      v = Xbfu[srow * 64 + (col >> 1)];
                else if (col < 256) v = Xbfu[drow * 64 + ((col - 128) >> 1)];
                else {
                    const float* p = ea + (size_t)e * 64 + (col - 256);
                    v = pack2bf(p[0], p[1]);
                }
                a.u[r] = v;
            }
#pragma unroll
            for (int tn = 0; tn < 8; ++tn) {
                Frag bf = load_frag_lds(ldsW1 + ((kt * 8 + tn) * 32 + lane) * 8);
                acc[tn] = wmma_bf16(a.v, bf.v, acc[tn]);
            }
        }

        // GELU -> bf16 -> per-wave LDS tile (D-layout -> A-layout transpose)
        unsigned short* m1h = (unsigned short*)(scr + 160);
#pragma unroll
        for (int tn = 0; tn < 8; ++tn) {
#pragma unroll
            for (int r = 0; r < 8; ++r) {
                int row = half * 8 + r;
                m1h[row * 128 + tn * 16 + M] = (unsigned short)f2bf1(gelu_exact(acc[tn][r]));
            }
        }
        lds_fence();

        // GEMM2: m = gelu(m1) (16x128) @ msg_w2 (128x128)
        v8f acc2[8];
#pragma unroll
        for (int tn = 0; tn < 8; ++tn) {
            float bb = b2[tn * 16 + M];
#pragma unroll
            for (int i = 0; i < 8; ++i) acc2[tn][i] = bb;
        }
        const unsigned int* m1u = scr + 160;
#pragma unroll
        for (int kt = 0; kt < 4; ++kt) {
            Frag a;
#pragma unroll
            for (int r = 0; r < 8; ++r) {
                const int col = kt * 32 + ((r >> 2) << 4) + (half << 3) + ((r & 3) << 1);
                a.u[r] = m1u[M * 64 + (col >> 1)];
            }
#pragma unroll
            for (int tn = 0; tn < 8; ++tn) {
                Frag bf = load_frag_lds(ldsW2 + ((kt * 8 + tn) * 32 + lane) * 8);
                acc2[tn] = wmma_bf16(a.v, bf.v, acc2[tn]);
            }
        }

        // attention scale (head == tn since HEAD_DIM==16) + scatter-add
#pragma unroll
        for (int tn = 0; tn < 8; ++tn) {
#pragma unroll
            for (int r = 0; r < 8; ++r) {
                int row = half * 8 + r;
                float an = __uint_as_float(scr[32 + row * 8 + tn]);
                int dn = (int)scr[16 + row];
                atomicAdd(agg + (size_t)dn * 128 + tn * 16 + M, acc2[tn][r] * an);
            }
        }
    }
}

// ---------------------------------------------------------------------------
// Node output MLP + residual + LayerNorm. Persistent, 16 nodes per wave-iter.
// LDS: out_w1 frags (16384 dw) | out_w2 frags (8192 dw) | 8 x 1024 dw scratch
// ---------------------------------------------------------------------------
__global__ void __launch_bounds__(256)
node_out_kernel(const float* __restrict__ nf, const unsigned int* __restrict__ Xbfu,
                const float* __restrict__ agg,
                const float* __restrict__ b1, const float* __restrict__ b2,
                const unsigned int* __restrict__ o1p, const unsigned int* __restrict__ o2p,
                const float* __restrict__ lng, const float* __restrict__ lnb,
                float* __restrict__ out) {
    extern __shared__ unsigned int smem[];
    unsigned int* ldsW1 = smem;
    unsigned int* ldsW2 = smem + 16384;
    const int tid = threadIdx.x;
    stage_lds(ldsW1, o1p, 16384, tid);
    stage_lds(ldsW2, o2p, 8192, tid);
    async_wait0();
    __syncthreads();

    const int wave = tid >> 5;
    const int lane = tid & 31;
    const int M = lane & 15;
    const int half = lane >> 4;
    unsigned int* scr = smem + 24576 + wave * 1024;
    const int nt = NN / 16;              // 625 tiles
    const int tstride = gridDim.x * 8;

    for (int t = blockIdx.x * 8 + wave; t < nt; t += tstride) {
        const int nb = t * 16;
        const int n = nb + M;

        // GEMM1: h1 = [x | agg] (16x256) @ out_w1 (256x128)
        v8f acc[8];
#pragma unroll
        for (int tn = 0; tn < 8; ++tn) {
            float bb = b1[tn * 16 + M];
#pragma unroll
            for (int i = 0; i < 8; ++i) acc[tn][i] = bb;
        }
#pragma unroll
        for (int kt = 0; kt < 8; ++kt) {
            Frag a;
#pragma unroll
            for (int r = 0; r < 8; ++r) {
                const int col = kt * 32 + ((r >> 2) << 4) + (half << 3) + ((r & 3) << 1);
                if (col < 128) a.u[r] = Xbfu[n * 64 + (col >> 1)];
                else {
                    const float* p = agg + (size_t)n * 128 + (col - 128);
                    a.u[r] = pack2bf(p[0], p[1]);
                }
            }
#pragma unroll
            for (int tn = 0; tn < 8; ++tn) {
                Frag bf = load_frag_lds(ldsW1 + ((kt * 8 + tn) * 32 + lane) * 8);
                acc[tn] = wmma_bf16(a.v, bf.v, acc[tn]);
            }
        }

        unsigned short* m1h = (unsigned short*)scr;
#pragma unroll
        for (int tn = 0; tn < 8; ++tn) {
#pragma unroll
            for (int r = 0; r < 8; ++r) {
                int row = half * 8 + r;
                m1h[row * 128 + tn * 16 + M] = (unsigned short)f2bf1(gelu_exact(acc[tn][r]));
            }
        }
        lds_fence();

        v8f acc2[8];
#pragma unroll
        for (int tn = 0; tn < 8; ++tn) {
            float bb = b2[tn * 16 + M];
#pragma unroll
            for (int i = 0; i < 8; ++i) acc2[tn][i] = bb;
        }
        const unsigned int* m1u = scr;
#pragma unroll
        for (int kt = 0; kt < 4; ++kt) {
            Frag a;
#pragma unroll
            for (int r = 0; r < 8; ++r) {
                const int col = kt * 32 + ((r >> 2) << 4) + (half << 3) + ((r & 3) << 1);
                a.u[r] = m1u[M * 64 + (col >> 1)];
            }
#pragma unroll
            for (int tn = 0; tn < 8; ++tn) {
                Frag bf = load_frag_lds(ldsW2 + ((kt * 8 + tn) * 32 + lane) * 8);
                acc2[tn] = wmma_bf16(a.v, bf.v, acc2[tn]);
            }
        }

        // residual + LayerNorm; row M' = half*8+r lives in the 16 lanes of this
        // half (one (L&15) slot x 8 tn each) -> 16-lane shfl_xor reductions.
        float gv[8], bv[8];
#pragma unroll
        for (int tn = 0; tn < 8; ++tn) { gv[tn] = lng[tn * 16 + M]; bv[tn] = lnb[tn * 16 + M]; }
#pragma unroll
        for (int r = 0; r < 8; ++r) {
            int n2 = nb + half * 8 + r;
            float vals[8];
            float s = 0.0f, q = 0.0f;
#pragma unroll
            for (int tn = 0; tn < 8; ++tn) {
                float v = acc2[tn][r] + nf[(size_t)n2 * 128 + tn * 16 + M];
                vals[tn] = v; s += v; q += v * v;
            }
#pragma unroll
            for (int m = 1; m < 16; m <<= 1) {
                s += __shfl_xor(s, m, 16);
                q += __shfl_xor(q, m, 16);
            }
            float mean = s * (1.0f / 128.0f);
            float var  = q * (1.0f / 128.0f) - mean * mean;
            float inv  = rsqrtf(var + 1e-5f);
#pragma unroll
            for (int tn = 0; tn < 8; ++tn)
                out[(size_t)n2 * 128 + tn * 16 + M] = (vals[tn] - mean) * inv * gv[tn] + bv[tn];
        }
    }
}

// ---------------------------------------------------------------------------
extern "C" void kernel_launch(void* const* d_in, const int* in_sizes, int n_in,
                              void* d_out, int out_size, void* d_ws, size_t ws_size,
                              hipStream_t stream) {
    (void)in_sizes; (void)n_in; (void)out_size; (void)ws_size;
    const float* nf  = (const float*)d_in[0];
    const float* ea  = (const float*)d_in[1];
    const float* mw1 = (const float*)d_in[2];
    const float* mb1 = (const float*)d_in[3];
    const float* mw2 = (const float*)d_in[4];
    const float* mb2 = (const float*)d_in[5];
    const float* qw  = (const float*)d_in[6];
    const float* qb  = (const float*)d_in[7];
    const float* kvw = (const float*)d_in[8];
    const float* kvb = (const float*)d_in[9];
    const float* ow1 = (const float*)d_in[10];
    const float* ob1 = (const float*)d_in[11];
    const float* ow2 = (const float*)d_in[12];
    const float* ob2 = (const float*)d_in[13];
    const float* lng = (const float*)d_in[14];
    const float* lnb = (const float*)d_in[15];
    const int*   ei  = (const int*)d_in[16];
    const int* srcI = ei;
    const int* dstI = ei + NE;
    float* out = (float*)d_out;

    char* ws = (char*)d_ws;
    size_t off = 0;
    auto take = [&](size_t bytes) -> char* {
        char* p = ws + off;
        off = (off + bytes + 255) & ~(size_t)255;
        return p;
    };
    int*          last  = (int*)take(NN * 4);
    float*        Q     = (float*)take((size_t)NN * 128 * 4);
    unsigned int* Xbfu  = (unsigned int*)take((size_t)NN * 64 * 4);
    float*        attnF = (float*)take(NN * 8 * 4);
    float*        attnN = (float*)take(NN * 8 * 4);
    float*        stats = (float*)take(64);
    float*        agg   = (float*)take((size_t)NN * 128 * 4);
    unsigned int* w1p   = (unsigned int*)take(80 * 256 * 4);
    unsigned int* w2p   = (unsigned int*)take(32 * 256 * 4);
    unsigned int* o1p   = (unsigned int*)take(64 * 256 * 4);
    unsigned int* o2p   = (unsigned int*)take(32 * 256 * 4);

    const int edgeLds = 38144 * 4;   // 149.0 KB (weights + per-wave scratch)
    const int nodeLds = 32768 * 4;   // 128.0 KB
    static bool attrDone = false;    // host-side attr set, not a device-state guard
    if (!attrDone) {
        (void)hipFuncSetAttribute((const void*)edge_msg_kernel,
                                  hipFuncAttributeMaxDynamicSharedMemorySize, edgeLds);
        (void)hipFuncSetAttribute((const void*)node_out_kernel,
                                  hipFuncAttributeMaxDynamicSharedMemorySize, nodeLds);
        attrDone = true;
    }

    init_kernel<<<2048, 256, 0, stream>>>(last, attnF, agg);
    xbf_kernel<<<(NN * 64 + 255) / 256, 256, 0, stream>>>(nf, Xbfu);
    pack_weight_kernel<<<80, 256, 0, stream>>>(mw1, w1p, 10);
    pack_weight_kernel<<<32, 256, 0, stream>>>(mw2, w2p, 4);
    pack_weight_kernel<<<64, 256, 0, stream>>>(ow1, o1p, 8);
    pack_weight_kernel<<<32, 256, 0, stream>>>(ow2, o2p, 4);
    last_kernel<<<(NE + 255) / 256, 256, 0, stream>>>(dstI, last);
    q_kernel<<<NN, 128, 0, stream>>>(nf, qw, qb, Q);
    attn_sel_kernel<<<NN, 128, 0, stream>>>(nf, ea, srcI, Q, kvw, kvb, last, attnF);
    softmax_stats_kernel<<<8, 256, 0, stream>>>(attnF, stats);
    attn_norm_kernel<<<(NN * 8 + 255) / 256, 256, 0, stream>>>(attnF, stats, attnN);
    // persistent grids: 500*8 waves x exactly 5 tiles = 20000 edge tiles
    edge_msg_kernel<<<500, 256, edgeLds, stream>>>(ea, srcI, dstI, Xbfu, attnN,
                                                   mb1, mb2, w1p, w2p, agg);
    node_out_kernel<<<40, 256, nodeLds, stream>>>(nf, Xbfu, agg, ob1, ob2,
                                                  o1p, o2p, lng, lnb, out);
}
